// FCN_12292196402087
// MI455X (gfx1250) — matrix operand order repaired and checked by reading, not compiled, not run on hardware
//
#include <hip/hip_runtime.h>

// ---------------------------------------------------------------------------
// 3-layer LSTM (H=32,32,3) over T timesteps on gfx1250 (MI455X).
// Single workgroup, 3 waves, software-pipelined across layers via LDS +
// workgroup split-barrier. Recurrent matvecs use v_wmma_f32_16x16x32_f16
// with weights resident in VGPR A-fragments (fp16) and f32 accumulation.
// Hidden states ride through LDS as packed f16 (no per-step cvt chains);
// nonlinearities use v_rcp_f32 / v_tanh_f32 instead of IEEE division.
// ---------------------------------------------------------------------------

typedef __attribute__((ext_vector_type(16))) _Float16     v16h;
typedef __attribute__((ext_vector_type(8)))  float        v8f;
typedef __attribute__((ext_vector_type(8)))  unsigned int v8u;

#define DEVINL __device__ __forceinline__

DEVINL float fsigm(float x) {
    // sigmoid with hardware rcp (v_rcp_f32) -- avoids v_div_scale chains
    return __builtin_amdgcn_rcpf(1.0f + __expf(-x));
}

#if __has_builtin(__builtin_amdgcn_tanhf)
DEVINL float ftanh_(float x) { return __builtin_amdgcn_tanhf(x); }   // v_tanh_f32
#else
DEVINL float ftanh_(float x) {
    float e = __expf(-2.0f * x);
    return (1.0f - e) * __builtin_amdgcn_rcpf(1.0f + e);
}
#endif

DEVINL v8f wmma_f16(v16h a, v16h b, v8f c) {
    // D = A(16x32 f16) x B(32x16 f16) + C(16x16 f32)
    return __builtin_amdgcn_wmma_f32_16x16x32_f16(
        /*neg_a=*/false, a, /*neg_b=*/false, b,
        /*c_mod=*/(short)0, c, /*reuse_a=*/false, /*reuse_b=*/false);
}

// Build a 16x32 fp16 A-fragment from row-major f32 weights.
// Layout (ISA 7.12.2, 16-bit A 16x32): lane m=l%16 holds row rowbase+m;
// VGPR v<4: K = 2v+{0,1} (+8 for lanes>=16); v>=4: K = 16+2(v-4)+{0,1} (+8).
DEVINL v16h load_a(const float* __restrict__ W, int rowbase, int ld,
                   int nrows, int ncols) {
    const int lane = threadIdx.x & 31;
    const int m    = lane & 15;
    const int hi8  = (lane >> 4) ? 8 : 0;
    const int r    = rowbase + m;
    v16h a;
#pragma unroll
    for (int v = 0; v < 8; ++v) {
#pragma unroll
        for (int s = 0; s < 2; ++s) {
            int k = ((v < 4) ? (2 * v) : (16 + 2 * (v - 4))) + hi8 + s;
            float val = (r < nrows && k < ncols) ? W[r * ld + k] : 0.0f;
            a[2 * v + s] = (_Float16)val;
        }
    }
    return a;
}

// Build a 32x16 fp16 B-fragment broadcasting vector h (packed f16 in LDS)
// over all 16 columns. VGPR v<4 -> K=2v,2v+1 (+8 for lanes>=16);
// v>=4 -> K=16+2(v-4)+{0,1} (+8). Pure dword LDS loads, zero conversions.
DEVINL v16h make_b(const _Float16* h) {
    const int lane = threadIdx.x & 31;
    const unsigned int* p =
        (const unsigned int*)(h + ((lane >> 4) ? 8 : 0));
    v8u u;
#pragma unroll
    for (int v = 0; v < 8; ++v) u[v] = p[(v < 4) ? v : (v + 4)];
    return __builtin_bit_cast(v16h, u);
}

// Scatter D (16x16 f32, all columns identical) to LDS gates[rowbase..+15].
// C/D layout: VGPR v -> row v (lanes 0-15) or row v+8 (lanes 16-31).
DEVINL void store_d(float* g, int rowbase, v8f d) {
    const int lane = threadIdx.x & 31;
    if ((lane & 15) == 0) {
        const int base = rowbase + ((lane >> 4) ? 8 : 0);
#pragma unroll
        for (int v = 0; v < 8; ++v) g[base + v] = d[v];
    }
}

__global__ __launch_bounds__(96, 1)
void lstm3_pipeline_kernel(
    const float* __restrict__ x,
    const float* __restrict__ wih0, const float* __restrict__ whh0,
    const float* __restrict__ bih0, const float* __restrict__ bhh0,
    const float* __restrict__ wih1, const float* __restrict__ whh1,
    const float* __restrict__ bih1, const float* __restrict__ bhh1,
    const float* __restrict__ pre_h0, const float* __restrict__ pre_c0,
    const float* __restrict__ wih2, const float* __restrict__ whh2,
    const float* __restrict__ bih2, const float* __restrict__ bhh2,
    const float* __restrict__ post_h0, const float* __restrict__ post_c0,
    float* __restrict__ out, int T)
{
    __shared__ __align__(16) _Float16 s_h1h[2][32]; // layer1 -> layer2 (f16)
    __shared__ __align__(16) _Float16 s_h2h[2][32]; // layer2 -> layer3 (f16)
    __shared__ __align__(16) _Float16 s_hsh[3][32]; // each wave's own h (f16)
    __shared__ float s_g[3][128];                   // per-wave gate scratch
    __shared__ __align__(16) float s_w0f[128];      // wih0 in C-fragment order
    __shared__ __align__(16) float s_b1f[128];      // b1   in C-fragment order

    const int tid  = threadIdx.x;
    const int lane = tid & 31;
    const int wave = tid >> 5;
    const int hi8  = (lane >> 4) ? 8 : 0;

    // ---- cooperative init: layer-1 input weights/bias, pre-swizzled to
    // C-fragment order: idx = hi*64 + q*8 + v  <->  row r = 16q + v + 8hi --
    for (int i = tid; i < 128; i += 96) {
        int q = (i >> 3) & 7, v = i & 7, hi = i >> 6;
        int r = 16 * q + v + 8 * hi;
        s_w0f[i] = wih0[r];
        s_b1f[i] = bih0[r] + bhh0[r];
    }

    // ---- per-wave recurrent state, weight A-fragments, bias C-fragments --
    float hreg = 0.0f, creg = 0.0f;
    v16h A[16];
    v8f  CB[8];
#pragma unroll
    for (int q = 0; q < 16; ++q) A[q] = v16h(0);
#pragma unroll
    for (int q = 0; q < 8; ++q)  CB[q] = v8f(0.0f);

    if (wave == 0) {
        hreg = pre_h0[lane];
        creg = pre_c0[lane];
#pragma unroll
        for (int q = 0; q < 8; ++q) A[q] = load_a(whh0, 16 * q, 32, 128, 32);
        s_hsh[0][lane] = (_Float16)hreg;
    } else if (wave == 1) {
        hreg = pre_h0[32 + lane];
        creg = pre_c0[32 + lane];
#pragma unroll
        for (int q = 0; q < 8; ++q) {
            A[q]     = load_a(whh1, 16 * q, 32, 128, 32);
            A[8 + q] = load_a(wih1, 16 * q, 32, 128, 32);
#pragma unroll
            for (int v = 0; v < 8; ++v) {
                int r = 16 * q + v + hi8;
                CB[q][v] = bih1[r] + bhh1[r];
            }
        }
        s_hsh[1][lane] = (_Float16)hreg;
    } else {
        if (lane < 3) { hreg = post_h0[lane]; creg = post_c0[lane]; }
        A[0] = load_a(wih2, 0, 32, 12, 32);  // 12x32 (rows 12..15 zero)
        A[1] = load_a(whh2, 0, 3, 12, 3);    // 12x3 padded to 16x32
#pragma unroll
        for (int v = 0; v < 8; ++v) {
            int r = v + hi8;
            CB[0][v] = (r < 12) ? (bih2[r] + bhh2[r]) : 0.0f;
        }
        s_hsh[2][lane] = (_Float16)hreg;     // lanes >=3 stay exactly 0 forever
    }
    __syncthreads();

    const float* w0p = &s_w0f[(lane >> 4) ? 64 : 0];
    const float* b1p = &s_b1f[(lane >> 4) ? 64 : 0];

    // ---- pipelined time loop: wave w handles step t-w --------------------
    for (int t = 0; t < T + 2; ++t) {
        if (wave == 0) {
            if (t < T) {
                if (lane == 0 && t + 64 < T) __builtin_prefetch(&x[t + 64], 0, 0);
                const float xn = x[t] * 0.04f;           // (x-0)/25
                v16h b = make_b(s_hsh[0]);
#pragma unroll
                for (int q = 0; q < 8; ++q) {
                    v8f c;
#pragma unroll
                    for (int v = 0; v < 8; ++v)
                        c[v] = fmaf(xn, w0p[q * 8 + v], b1p[q * 8 + v]);
                    v8f d = wmma_f16(A[q], b, c);        // += Whh0 @ h
                    store_d(s_g[0], 16 * q, d);
                }
                float gi = s_g[0][lane], gf = s_g[0][32 + lane];
                float gg = s_g[0][64 + lane], go = s_g[0][96 + lane];
                creg = fsigm(gf) * creg + fsigm(gi) * ftanh_(gg);
                hreg = fsigm(go) * ftanh_(creg);
                _Float16 hh = (_Float16)hreg;
                s_hsh[0][lane]     = hh;
                s_h1h[t & 1][lane] = hh;
            }
        } else if (wave == 1) {
            const int st = t - 1;
            if (st >= 0 && st < T) {
                v16h bh = make_b(s_hsh[1]);
                v16h bx = make_b(s_h1h[st & 1]);
#pragma unroll
                for (int q = 0; q < 8; ++q) {
                    v8f d = wmma_f16(A[q], bh, CB[q]);   // b2 + Whh1 @ h_prev
                    d      = wmma_f16(A[8 + q], bx, d);  //    + Wih1 @ h1_t
                    store_d(s_g[1], 16 * q, d);
                }
                float gi = s_g[1][lane], gf = s_g[1][32 + lane];
                float gg = s_g[1][64 + lane], go = s_g[1][96 + lane];
                creg = fsigm(gf) * creg + fsigm(gi) * ftanh_(gg);
                hreg = fsigm(go) * ftanh_(creg);
                _Float16 hh = (_Float16)hreg;
                s_hsh[1][lane]      = hh;
                s_h2h[st & 1][lane] = hh;
            }
        } else if (wave == 2) {
            const int st = t - 2;
            if (st >= 0 && st < T) {
                v16h bh = make_b(s_hsh[2]);              // h3 (lanes>=3 are 0)
                v16h bx = make_b(s_h2h[st & 1]);
                v8f d = wmma_f16(A[1], bh, CB[0]);       // b3 + Whh2 @ h3_prev
                d      = wmma_f16(A[0], bx, d);          //    + Wih2 @ h2_t
                store_d(s_g[2], 0, d);
                if (lane < 3) {
                    float gi = s_g[2][lane],     gf = s_g[2][3 + lane];
                    float gg = s_g[2][6 + lane], go = s_g[2][9 + lane];
                    creg = fsigm(gf) * creg + fsigm(gi) * ftanh_(gg);
                    hreg = fsigm(go) * ftanh_(creg);
                    s_hsh[2][lane] = (_Float16)hreg;
                    out[st * 3 + lane] = hreg;
                }
            }
        }
        __syncthreads();   // pipeline tick (s_barrier_signal/wait on gfx1250)
    }
}

extern "C" void kernel_launch(void* const* d_in, const int* in_sizes, int n_in,
                              void* d_out, int out_size, void* d_ws, size_t ws_size,
                              hipStream_t stream) {
    (void)n_in; (void)out_size; (void)d_ws; (void)ws_size;
    const float* x       = (const float*)d_in[0];
    const float* wih0    = (const float*)d_in[1];
    const float* whh0    = (const float*)d_in[2];
    const float* bih0    = (const float*)d_in[3];
    const float* bhh0    = (const float*)d_in[4];
    const float* wih1    = (const float*)d_in[5];
    const float* whh1    = (const float*)d_in[6];
    const float* bih1    = (const float*)d_in[7];
    const float* bhh1    = (const float*)d_in[8];
    const float* pre_h0  = (const float*)d_in[9];
    const float* pre_c0  = (const float*)d_in[10];
    const float* wih2    = (const float*)d_in[11];
    const float* whh2    = (const float*)d_in[12];
    const float* bih2    = (const float*)d_in[13];
    const float* bhh2    = (const float*)d_in[14];
    const float* post_h0 = (const float*)d_in[15];
    const float* post_c0 = (const float*)d_in[16];

    const int T = in_sizes[0];

    lstm3_pipeline_kernel<<<1, 96, 0, stream>>>(
        x, wih0, whh0, bih0, bhh0, wih1, whh1, bih1, bhh1,
        pre_h0, pre_c0, wih2, whh2, bih2, bhh2, post_h0, post_c0,
        (float*)d_out, T);
}
